// GAT_21165598834729
// MI455X (gfx1250) — matrix operand order repaired and checked by reading, not compiled, not run on hardware
//
#include <hip/hip_runtime.h>
#include <hip/hip_bf16.h>
#include <math.h>

// ---------------------------------------------------------------------------
// GAT (3-layer) for MI455X / gfx1250.
//   - GEMMs via v_wmma_f32_16x16x32_f16 (f16 inputs, f32 accumulate)
//   - LDS staging via async global->LDS (ASYNCcnt double-buffering) when the
//     toolchain exposes the builtin; clamped-row sync staging otherwise.
//   - edge softmax via 3-pass atomics (node features = 41MB, L2-resident)
// ---------------------------------------------------------------------------

typedef __attribute__((ext_vector_type(16))) _Float16 v16h;
typedef __attribute__((ext_vector_type(8)))  _Float16 v8h;
typedef __attribute__((ext_vector_type(8)))  float    v8f;

#define TILE_M  128
#define TILE_N  64
#define LDA_PAD 40   // halves per LDS row (32 data + 8 pad) -> conflict-free 16-lane reads

#if defined(__AMDGCN__) && __has_builtin(__builtin_amdgcn_global_load_async_to_lds_b128)
#define GAT_ASYNC_LDS 1
#endif

#ifdef GAT_ASYNC_LDS
#define AS1 __attribute__((address_space(1)))
#define AS3 __attribute__((address_space(3)))
typedef int v4i_vs __attribute__((vector_size(4 * sizeof(int))));  // matches builtin pointee
__device__ __forceinline__ void async_b128(const _Float16* g, _Float16* l) {
  __builtin_amdgcn_global_load_async_to_lds_b128((AS1 v4i_vs*)g, (AS3 v4i_vs*)l, 0, 0);
}
__device__ __forceinline__ void wait_async_le3() {
#if __has_builtin(__builtin_amdgcn_s_wait_asynccnt)
  __builtin_amdgcn_s_wait_asynccnt(3);
#else
  asm volatile("s_wait_asynccnt 0x3" ::: "memory");
#endif
}
__device__ __forceinline__ void wait_async_le0() {
#if __has_builtin(__builtin_amdgcn_s_wait_asynccnt)
  __builtin_amdgcn_s_wait_asynccnt(0);
#else
  asm volatile("s_wait_asynccnt 0x0" ::: "memory");
#endif
}
#endif

// ---------------------------- helpers --------------------------------------

__device__ __forceinline__ void atomicMaxF(float* addr, float val) {
  // signed-max for >=0, unsigned-min for <0 gives a correct float max
  if (val >= 0.0f) atomicMax((int*)addr, __float_as_int(val));
  else             atomicMin((unsigned int*)addr, __float_as_uint(val));
}

// ---------------------------- conversion -----------------------------------

__global__ __launch_bounds__(256) void k_convert_f16(const float* __restrict__ x,
                                                     _Float16* __restrict__ xh, int n) {
  int i = blockIdx.x * 256 + threadIdx.x;
  if (i < n) xh[i] = (_Float16)x[i];
}

// W [K, Nout] row-major f32  ->  WT [Nout, K] row-major f16
__global__ __launch_bounds__(256) void k_convert_WT(const float* __restrict__ W,
                                                    _Float16* __restrict__ WT,
                                                    int K, int Nout) {
  int i = blockIdx.x * 256 + threadIdx.x;
  if (i >= K * Nout) return;
  int k = i / Nout, n = i % Nout;
  WT[(size_t)n * K + k] = (_Float16)W[i];
}

__global__ __launch_bounds__(256) void k_fill(float* __restrict__ p, float v, int n) {
  int i = blockIdx.x * 256 + threadIdx.x;
  if (i < n) p[i] = v;
}

// ---------------------------- WMMA GEMM ------------------------------------
// C[M,Nout](f32) = A[M,K](f16) x BT[Nout,K](f16)^T
// Block: 256 threads = 8 waves; block tile 128x64; wave tile 16x64.
// Double-buffered LDS; async global->LDS pipeline with counted ASYNCcnt waits.
__global__ __launch_bounds__(256) void k_gemm_wmma(const _Float16* __restrict__ A,
                                                   const _Float16* __restrict__ BT,
                                                   float* __restrict__ C,
                                                   int M, int K, int Nout) {
  __shared__ _Float16 As[2][TILE_M * LDA_PAD];
  __shared__ _Float16 Bs[2][TILE_N * LDA_PAD];

  const int tid  = threadIdx.x;
  const int row0 = blockIdx.x * TILE_M;
  const int col0 = blockIdx.y * TILE_N;
  const int wave = tid >> 5;
  const int lane = tid & 31;
  const int ml   = lane & 15;   // M (A frag), N (B frag / D)
  const int gl   = lane >> 4;   // lane-half select
  const int rm   = wave * 16;   // wave's row offset within block tile

  v8f zero = {};
  v8f acc[4];
#pragma unroll
  for (int i = 0; i < 4; i++) acc[i] = zero;

  // staging assignments (all unconditional: OOB rows are CLAMPED, their garbage
  // only reaches accumulators whose stores are row-guarded)
  const int ar = tid >> 1;            // A: 128 rows, 2 threads/row
  const int ao = (tid & 1) * 16;      // 16 halves (2 x b128) each
  const int br = tid >> 2;            // B: 64 rows, 4 threads/row
  const int bo = (tid & 3) * 8;       // 8 halves (1 x b128) each

  int arow = row0 + ar;
  if (arow >= M) arow = M - 1;        // clamp instead of branch
  const _Float16* aSrc = A  + (size_t)arow * K + ao;
  const _Float16* bSrc = BT + (size_t)(col0 + br) * K + bo;
  _Float16* aDst0 = &As[0][ar * LDA_PAD + ao];
  _Float16* aDst1 = &As[1][ar * LDA_PAD + ao];
  _Float16* bDst0 = &Bs[0][br * LDA_PAD + bo];
  _Float16* bDst1 = &Bs[1][br * LDA_PAD + bo];

  auto stage = [&](int buf, int k0) {
    _Float16* ad = buf ? aDst1 : aDst0;
    _Float16* bd = buf ? bDst1 : bDst0;
#ifdef GAT_ASYNC_LDS
    async_b128(aSrc + k0,     ad);       // 3 async ops per wave per tile
    async_b128(aSrc + k0 + 8, ad + 8);
    async_b128(bSrc + k0,     bd);
#else
    *(uint4*)ad       = *(const uint4*)(aSrc + k0);
    *(uint4*)(ad + 8) = *(const uint4*)(aSrc + k0 + 8);
    *(uint4*)bd       = *(const uint4*)(bSrc + k0);
#endif
  };

  const int nIter = K >> 5;           // K/32
  stage(0, 0);

  for (int it = 0; it < nIter; it++) {
    const int buf = it & 1;
    const bool more = (it + 1) < nIter;
    if (more) stage(buf ^ 1, (it + 1) * 32);
#ifdef GAT_ASYNC_LDS
    if (more) wait_async_le3();       // <=3 newest outstanding -> tile `it` landed
    else      wait_async_le0();
#endif
    __syncthreads();

    // ---- A fragment (ISA 16-bit A 16x32 layout): lane-half gl selects octets
    v16h af;
    {
      const _Float16* ab = &As[buf][(rm + ml) * LDA_PAD];
      v8h lo = *(const v8h*)(ab + 8 * gl);        // K = 8g..8g+7
      v8h hi = *(const v8h*)(ab + 16 + 8 * gl);   // K = 16+8g..+7
#pragma unroll
      for (int i = 0; i < 8; i++) { af[i] = lo[i]; af[i + 8] = hi[i]; }
    }
    // ---- preload all 4 B fragments, then issue WMMAs back-to-back ----
    v16h bf[4];
#pragma unroll
    for (int nt = 0; nt < 4; nt++) {
      const _Float16* bb = &Bs[buf][(nt * 16 + ml) * LDA_PAD];
      v8h blo = *(const v8h*)(bb + 8 * gl);
      v8h bhi = *(const v8h*)(bb + 16 + 8 * gl);
#pragma unroll
      for (int i = 0; i < 8; i++) { bf[nt][i] = blo[i]; bf[nt][i + 8] = bhi[i]; }
    }
#pragma unroll
    for (int nt = 0; nt < 4; nt++) {
      acc[nt] = __builtin_amdgcn_wmma_f32_16x16x32_f16(
          /*neg_a=*/false, af, /*neg_b=*/false, bf[nt],
          /*c_mod=*/(short)0, acc[nt], /*reuse_a=*/false, /*reuse_b=*/false);
    }
    __syncthreads();   // all waves done reading buf before it is refilled
  }

  // ---- store D (ISA C/D layout: VGPR r -> M = r + 8*gl, N = ml) ----
#pragma unroll
  for (int nt = 0; nt < 4; nt++) {
#pragma unroll
    for (int r = 0; r < 8; r++) {
      int grow = row0 + rm + r + 8 * gl;
      if (grow < M)
        C[(size_t)grow * Nout + col0 + nt * 16 + ml] = acc[nt][r];
    }
  }
}

// ---------------------------- attention ------------------------------------

// a_src[n,h] = sum_c h[n,h,c]*att_src[h,c] ; likewise a_dst
__global__ __launch_bounds__(256) void k_att_scores(const float* __restrict__ h,
                                                    const float* __restrict__ att_s,
                                                    const float* __restrict__ att_d,
                                                    float* __restrict__ aS,
                                                    float* __restrict__ aD,
                                                    int n_nodes, int H, int C) {
  int idx = blockIdx.x * 256 + threadIdx.x;
  if (idx >= n_nodes * H) return;
  int node = idx / H, hd = idx % H;
  const float* hp = h + (size_t)node * H * C + (size_t)hd * C;
  const float* sp = att_s + (size_t)hd * C;
  const float* dp = att_d + (size_t)hd * C;
  float accs = 0.f, accd = 0.f;
  for (int c = 0; c < C; c++) {
    float v = hp[c];
    accs += v * sp[c];
    accd += v * dp[c];
  }
  aS[idx] = accs;
  aD[idx] = accd;
}

// pass1: e = leaky_relu(a_src[src]+a_dst[dst]); store; atomic segment max per dst
__global__ __launch_bounds__(256) void k_edge_pass1(const long long* __restrict__ ei,
                                                    const float* __restrict__ aS,
                                                    const float* __restrict__ aD,
                                                    float* __restrict__ eB,
                                                    float* __restrict__ mB,
                                                    int E, int Etot, int H) {
  int idx = blockIdx.x * 256 + threadIdx.x;
  if (idx >= Etot * H) return;
  int e = idx / H, hd = idx % H;
  int s, d;
  if (e < E) { s = (int)ei[e]; d = (int)ei[E + e]; }
  else       { s = d = e - E; }                    // appended self loops
  float v = aS[s * H + hd] + aD[d * H + hd];
  v = v > 0.f ? v : 0.2f * v;                      // leaky_relu, slope 0.2
  eB[idx] = v;
  atomicMaxF(&mB[d * H + hd], v);
}

// pass2: ex = exp(e - m[dst]); store; atomic segment sum -> denom
__global__ __launch_bounds__(256) void k_edge_pass2(const long long* __restrict__ ei,
                                                    float* __restrict__ eB,
                                                    const float* __restrict__ mB,
                                                    float* __restrict__ dB,
                                                    int E, int Etot, int H) {
  int idx = blockIdx.x * 256 + threadIdx.x;
  if (idx >= Etot * H) return;
  int e = idx / H, hd = idx % H;
  int d = (e < E) ? (int)ei[E + e] : (e - E);
  float ex = expf(eB[idx] - mB[d * H + hd]);
  eB[idx] = ex;
  atomicAdd(&dB[d * H + hd], ex);
}

// pass3: out[dst, f] += h[src, f] * ex/(denom[dst]+eps)   (f = head*C + c)
__global__ __launch_bounds__(256) void k_edge_scatter(const long long* __restrict__ ei,
                                                      const float* __restrict__ h,
                                                      const float* __restrict__ eB,
                                                      const float* __restrict__ dB,
                                                      float* __restrict__ outb,
                                                      int E, int Etot, int H, int C) {
  long long t = (long long)blockIdx.x * 256 + threadIdx.x;
  int F = H * C;
  if (t >= (long long)Etot * F) return;
  int e  = (int)(t / F);
  int f  = (int)(t % F);
  int hd = f / C;
  int s, d;
  if (e < E) { s = (int)ei[e]; d = (int)ei[E + e]; }
  else       { s = d = e - E; }
  float alpha = eB[(size_t)e * H + hd] / (dB[(size_t)d * H + hd] + 1e-16f);
  atomicAdd(&outb[(size_t)d * F + f], h[(size_t)s * F + f] * alpha);
}

// bias + ELU + f16 convert (feeds next layer's GEMM)
__global__ __launch_bounds__(256) void k_bias_elu_f16(const float* __restrict__ agg,
                                                      const float* __restrict__ bias,
                                                      _Float16* __restrict__ xh,
                                                      int n, int F) {
  int i = blockIdx.x * 256 + threadIdx.x;
  if (i >= n) return;
  float v = agg[i] + bias[i % F];
  v = v > 0.f ? v : expm1f(v);   // elu, alpha=1
  xh[i] = (_Float16)v;
}

// layer-3 epilogue: + b3, then log_softmax over 64 classes
__global__ __launch_bounds__(256) void k_finalize(const float* __restrict__ agg,
                                                  const float* __restrict__ b,
                                                  float* __restrict__ out,
                                                  int n_nodes, int C) {
  int node = blockIdx.x * 256 + threadIdx.x;
  if (node >= n_nodes) return;
  const float* row = agg + (size_t)node * C;
  float mx = -INFINITY;
  for (int c = 0; c < C; c++) mx = fmaxf(mx, row[c] + b[c]);
  float s = 0.f;
  for (int c = 0; c < C; c++) s += expf(row[c] + b[c] - mx);
  float lse = mx + logf(s);
  for (int c = 0; c < C; c++) out[(size_t)node * C + c] = row[c] + b[c] - lse;
}

// ---------------------------------------------------------------------------

extern "C" void kernel_launch(void* const* d_in, const int* in_sizes, int n_in,
                              void* d_out, int out_size, void* d_ws, size_t ws_size,
                              hipStream_t stream) {
  (void)n_in; (void)out_size; (void)ws_size;
  const int INW = 512, F = 512, HEADS = 8, HID = 64, OUTC = 64;
  const int N    = in_sizes[0] / INW;
  const int E    = in_sizes[1] / 2;
  const int Etot = E + N;

  const float*     x  = (const float*)d_in[0];
  const long long* ei = (const long long*)d_in[1];   // int64 [2,E]
  const float* Wm[3]  = {(const float*)d_in[2], (const float*)d_in[6],  (const float*)d_in[10]};
  const float* As_[3] = {(const float*)d_in[3], (const float*)d_in[7],  (const float*)d_in[11]};
  const float* Ad_[3] = {(const float*)d_in[4], (const float*)d_in[8],  (const float*)d_in[12]};
  const float* Bs_[3] = {(const float*)d_in[5], (const float*)d_in[9],  (const float*)d_in[13]};

  // workspace carve (256B aligned)
  char* wp = (char*)d_ws;
  auto carve = [&](size_t bytes) -> void* {
    void* p = (void*)wp;
    wp += (bytes + 255) & ~(size_t)255;
    return p;
  };
  float*    bufH  = (float*)   carve((size_t)N * F * 4);        // GEMM out (h)
  float*    bufAg = (float*)   carve((size_t)N * F * 4);        // aggregation out
  _Float16* xh    = (_Float16*)carve((size_t)N * F * 2);        // f16 GEMM input
  _Float16* WT    = (_Float16*)carve((size_t)F * F * 2);        // f16 transposed W
  float*    aS    = (float*)   carve((size_t)N * HEADS * 4);
  float*    aD    = (float*)   carve((size_t)N * HEADS * 4);
  float*    mB    = (float*)   carve((size_t)N * HEADS * 4);
  float*    dB    = (float*)   carve((size_t)N * HEADS * 4);
  float*    eB    = (float*)   carve((size_t)Etot * HEADS * 4);

  auto nb = [](long long n) { return (unsigned)((n + 255) / 256); };

  k_convert_f16<<<nb((long long)N * INW), 256, 0, stream>>>(x, xh, N * INW);

  for (int layer = 0; layer < 3; layer++) {
    const int H  = (layer == 2) ? 1 : HEADS;
    const int C  = (layer == 2) ? OUTC : HID;
    const int Fo = H * C;       // 512, 512, 64
    const int Kd = F;           // input dim is 512 for all layers

    k_convert_WT<<<nb((long long)Kd * Fo), 256, 0, stream>>>(Wm[layer], WT, Kd, Fo);

    dim3 gg((N + TILE_M - 1) / TILE_M, Fo / TILE_N);
    k_gemm_wmma<<<gg, 256, 0, stream>>>(xh, WT, bufH, N, Kd, Fo);

    k_att_scores<<<nb((long long)N * H), 256, 0, stream>>>(bufH, As_[layer], Ad_[layer],
                                                           aS, aD, N, H, C);

    k_fill<<<nb((long long)N * H), 256, 0, stream>>>(mB, -INFINITY, N * H);
    k_fill<<<nb((long long)N * H), 256, 0, stream>>>(dB, 0.0f, N * H);
    k_fill<<<nb((long long)N * Fo), 256, 0, stream>>>(bufAg, 0.0f, N * Fo);

    k_edge_pass1<<<nb((long long)Etot * H), 256, 0, stream>>>(ei, aS, aD, eB, mB, E, Etot, H);
    k_edge_pass2<<<nb((long long)Etot * H), 256, 0, stream>>>(ei, eB, mB, dB, E, Etot, H);
    k_edge_scatter<<<nb((long long)Etot * Fo), 256, 0, stream>>>(ei, bufH, eB, dB, bufAg,
                                                                 E, Etot, H, C);

    if (layer < 2) {
      k_bias_elu_f16<<<nb((long long)N * Fo), 256, 0, stream>>>(bufAg, Bs_[layer], xh,
                                                                N * Fo, Fo);
    } else {
      k_finalize<<<nb(N), 256, 0, stream>>>(bufAg, Bs_[layer], (float*)d_out, N, C);
    }
  }
}
